// ClusterAttentionAE_76785425318473
// MI455X (gfx1250) — matrix operand order repaired and checked by reading, not compiled, not run on hardware
//
#include <hip/hip_runtime.h>
#include <math.h>

#define DIN 128
#define DH 8
#define ATT_SLOPE 0.2f
#define ACT_SLOPE 0.01f

typedef __attribute__((ext_vector_type(16))) _Float16 v16h;
typedef __attribute__((ext_vector_type(8)))  float    v8f;

__device__ __forceinline__ float leaky(float v, float s) { return v > 0.f ? v : s * v; }

// float atomic max via signed/unsigned int ordering trick (init with -inf)
__device__ __forceinline__ void atomicMaxF(float* addr, float val) {
  if (val >= 0.f) atomicMax((int*)addr, __float_as_int(val));
  else            atomicMin((unsigned int*)addr, __float_as_uint(val));
}

// ---------------- init ----------------
__global__ void k_init(float* __restrict__ agg2, const float* __restrict__ dec_b,
                       float* __restrict__ agg1, const float* __restrict__ enc_b,
                       float* __restrict__ m, float* __restrict__ den,
                       int* __restrict__ maskflag, float* __restrict__ acc, int N) {
  int i = blockIdx.x * blockDim.x + threadIdx.x;
  int tot = N * DIN;
  if (i >= tot) return;
  agg2[i] = dec_b[i & (DIN - 1)];
  if (i < N * DH) agg1[i] = enc_b[i & (DH - 1)];
  if (i < N) { m[i] = -__builtin_huge_valf(); den[i] = 0.f; maskflag[i] = 0; }
  if (i == 0) acc[0] = 0.f;
}

__global__ void k_scatter_mask(const int* __restrict__ mask_nodes,
                               int* __restrict__ maskflag, int NM) {
  int i = blockIdx.x * blockDim.x + threadIdx.x;
  if (i < NM) maskflag[mask_nodes[i]] = 1;
}

__global__ void k_reinit(float* __restrict__ m, float* __restrict__ den, int N) {
  int i = blockIdx.x * blockDim.x + threadIdx.x;
  if (i < N) { m[i] = -__builtin_huge_valf(); den[i] = 0.f; }
}

// ---------------- encoder GEMM: h_enc = use_x @ enc_W^T via WMMA ----------------
// one wave per 16-node tile; 4 x v_wmma_f32_16x16x32_f16 along K=128
__global__ void k_enc_gemm(const float* __restrict__ x, const float* __restrict__ token,
                           const int* __restrict__ maskflag, const float* __restrict__ encW,
                           float* __restrict__ henc, int N, int NT) {
  int gid  = blockIdx.x * blockDim.x + threadIdx.x;
  int tile = gid >> 5;
  int lane = threadIdx.x & 31;
  if (tile >= NT) return;                       // wave-uniform exit (EXEC all-ones for WMMA)

  int row    = lane & 15;
  int node   = tile * 16 + row;
  int nodec  = node < N ? node : N - 1;
  int kbaseA = (lane < 16) ? 0 : 8;             // unused; A is loaded with kbase2 like B
  int c      = lane & 15;                       // B/D column
  int kbase2 = (lane < 16) ? 0 : 16;            // contiguous k = 16h + s, used for A and B
  const float* xr = maskflag[nodec] ? token : (x + nodec * DIN);

  v8f acc = {};
#pragma unroll
  for (int t = 0; t < 4; ++t) {
    int k0 = t * 32;
    v16h a, b;
#pragma unroll
    for (int p = 0; p < 8; ++p) {
      int K = kbase2 + 2 * p;
      float2 f2 = *(const float2*)(xr + k0 + K);
      a[2 * p]     = (_Float16)f2.x;
      a[2 * p + 1] = (_Float16)f2.y;
    }
#pragma unroll
    for (int i = 0; i < 16; ++i) b[i] = (_Float16)0.f;
    if (c < DH) {
      const float* wr = encW + c * DIN + k0 + kbase2;
#pragma unroll
      for (int i = 0; i < 16; i += 4) {
        float4 f = *(const float4*)(wr + i);
        b[i] = (_Float16)f.x; b[i + 1] = (_Float16)f.y;
        b[i + 2] = (_Float16)f.z; b[i + 3] = (_Float16)f.w;
      }
    }
    acc = __builtin_amdgcn_wmma_f32_16x16x32_f16(false, a, false, b, (short)0, acc, false, false);
  }
  // D layout: VGPR v -> row M = v + (lane<16?0:8), column = lane&15
  int mofs = (lane < 16) ? 0 : 8;
  if (c < DH) {
#pragma unroll
    for (int v = 0; v < 8; ++v) {
      int r = tile * 16 + v + mofs;
      if (r < N) henc[r * DH + c] = acc[v];
    }
  }
}

// ---------------- per-node attention scalars (encoder) ----------------
__global__ void k_enc_alpha(const float* __restrict__ h, const float* __restrict__ a_src,
                            const float* __restrict__ a_dst, float* __restrict__ asrc,
                            float* __restrict__ adst, int N) {
  int n = blockIdx.x * blockDim.x + threadIdx.x;
  if (n >= N) return;
  float4 h0 = *(const float4*)(h + n * DH);
  float4 h1 = *(const float4*)(h + n * DH + 4);
  asrc[n] = h0.x*a_src[0] + h0.y*a_src[1] + h0.z*a_src[2] + h0.w*a_src[3]
          + h1.x*a_src[4] + h1.y*a_src[5] + h1.z*a_src[6] + h1.w*a_src[7];
  adst[n] = h0.x*a_dst[0] + h0.y*a_dst[1] + h0.z*a_dst[2] + h0.w*a_dst[3]
          + h1.x*a_dst[4] + h1.y*a_dst[5] + h1.z*a_dst[6] + h1.w*a_dst[7];
}

// ---------------- edge softmax passes ----------------
__global__ void k_edge_logits(const int* __restrict__ ei, const float* __restrict__ asrc,
                              const float* __restrict__ adst, float* __restrict__ m,
                              float* __restrict__ ebuf, int E, int ET) {
  int e = blockIdx.x * blockDim.x + threadIdx.x;
  if (e >= ET) return;
  int s, d;
  if (e < E) { s = ei[e]; d = ei[E + e]; } else { s = d = e - E; }
  float v = leaky(asrc[s] + adst[d], ATT_SLOPE);
  ebuf[e] = v;
  atomicMaxF(&m[d], v);
}

__global__ void k_edge_exp(const int* __restrict__ ei, const float* __restrict__ m,
                           float* __restrict__ den, float* __restrict__ ebuf, int E, int ET) {
  int e = blockIdx.x * blockDim.x + threadIdx.x;
  if (e >= ET) return;
  int d = (e < E) ? ei[E + e] : (e - E);
  float ex = __expf(ebuf[e] - m[d]);
  ebuf[e] = ex;
  atomicAdd(&den[d], ex);
}

__global__ void k_edge_agg_enc(const int* __restrict__ ei, const float* __restrict__ henc,
                               const float* __restrict__ ebuf, const float* __restrict__ den,
                               float* __restrict__ agg1, int E, int ET) {
  int e = blockIdx.x * blockDim.x + threadIdx.x;
  if (e >= ET) return;
  int s, d;
  if (e < E) { s = ei[e]; d = ei[E + e]; } else { s = d = e - E; }
  float alpha = ebuf[e] / den[d];
  float4 h0 = *(const float4*)(henc + s * DH);
  float4 h1 = *(const float4*)(henc + s * DH + 4);
  float* ad = agg1 + d * DH;
  atomicAdd(ad + 0, alpha * h0.x); atomicAdd(ad + 1, alpha * h0.y);
  atomicAdd(ad + 2, alpha * h0.z); atomicAdd(ad + 3, alpha * h0.w);
  atomicAdd(ad + 4, alpha * h1.x); atomicAdd(ad + 5, alpha * h1.y);
  atomicAdd(ad + 6, alpha * h1.z); atomicAdd(ad + 7, alpha * h1.w);
}

// ---------------- encoder activation: enc_rep (in-place + output) ----------------
__global__ void k_enc_act(float* __restrict__ agg1, float* __restrict__ out, int tot) {
  int i = blockIdx.x * blockDim.x + threadIdx.x;
  if (i >= tot) return;
  float r = leaky(agg1[i], ACT_SLOPE);
  agg1[i] = r;
  out[i] = r;
}

// ---------------- fold decoder attention vectors: w = dec_W^T a ----------------
__global__ void k_fold(const float* __restrict__ decW, const float* __restrict__ a_src,
                       const float* __restrict__ a_dst, float* __restrict__ wsrc,
                       float* __restrict__ wdst) {
  int k = threadIdx.x;
  if (k < DH) {
    float s = 0.f;
    for (int j = 0; j < DIN; ++j) s += decW[j * DH + k] * a_src[j];
    wsrc[k] = s;
  } else if (k < 2 * DH) {
    int kk = k - DH;
    float s = 0.f;
    for (int j = 0; j < DIN; ++j) s += decW[j * DH + kk] * a_dst[j];
    wdst[kk] = s;
  }
}

// ---------------- e2d projection + re-mask + folded decoder alphas ----------------
__global__ void k_e2d(const float* __restrict__ enc_rep, const float* __restrict__ e2dW,
                      const int* __restrict__ maskflag, const float* __restrict__ wsrc,
                      const float* __restrict__ wdst, float* __restrict__ rep,
                      float* __restrict__ asrc2, float* __restrict__ adst2, int N) {
  int n = blockIdx.x * blockDim.x + threadIdx.x;
  if (n >= N) return;
  float h[DH];
  const float* hr = enc_rep + n * DH;
#pragma unroll
  for (int k = 0; k < DH; ++k) h[k] = hr[k];
  bool mk = maskflag[n] != 0;
  float r[DH];
#pragma unroll
  for (int j = 0; j < DH; ++j) {
    float s = 0.f;
#pragma unroll
    for (int k = 0; k < DH; ++k) s += h[k] * e2dW[j * DH + k];
    r[j] = mk ? 0.f : s;
  }
  float ss = 0.f, sd = 0.f;
#pragma unroll
  for (int k = 0; k < DH; ++k) { ss += r[k] * wsrc[k]; sd += r[k] * wdst[k]; }
#pragma unroll
  for (int k = 0; k < DH; ++k) rep[n * DH + k] = r[k];
  asrc2[n] = ss;
  adst2[n] = sd;
}

// ---------------- decoder aggregation: h_dec computed on the fly ----------------
// one wave per edge; dec_W staged in LDS; lane handles 4 of the 128 outputs
__global__ void k_edge_agg_dec(const int* __restrict__ ei, const float* __restrict__ rep,
                               const float* __restrict__ decW, const float* __restrict__ ebuf,
                               const float* __restrict__ den, float* __restrict__ agg2,
                               int E, int ET) {
  __shared__ float Wl[DIN * DH];
  for (int i = threadIdx.x; i < DIN * DH; i += blockDim.x) Wl[i] = decW[i];
  __syncthreads();

  int gid  = blockIdx.x * blockDim.x + threadIdx.x;
  int e    = gid >> 5;
  int lane = threadIdx.x & 31;
  if (e >= ET) return;
  int s, d;
  if (e < E) { s = ei[e]; d = ei[E + e]; } else { s = d = e - E; }
  float alpha = ebuf[e] / den[d];
  float4 r0 = *(const float4*)(rep + s * DH);
  float4 r1 = *(const float4*)(rep + s * DH + 4);
  int j0 = lane * 4;
#pragma unroll
  for (int jj = 0; jj < 4; ++jj) {
    int j = j0 + jj;
    const float* wr = Wl + j * DH;
    float hh = r0.x*wr[0] + r0.y*wr[1] + r0.z*wr[2] + r0.w*wr[3]
             + r1.x*wr[4] + r1.y*wr[5] + r1.z*wr[6] + r1.w*wr[7];
    atomicAdd(&agg2[d * DIN + j], alpha * hh);
  }
}

// ---------------- cosine loss over masked nodes (wave per node) ----------------
__global__ void k_loss(const int* __restrict__ mask_nodes, const float* __restrict__ x,
                       const float* __restrict__ agg2, float* __restrict__ acc, int NM) {
  int gid  = blockIdx.x * blockDim.x + threadIdx.x;
  int w    = gid >> 5;
  int lane = threadIdx.x & 31;
  if (w >= NM) return;
  int node = mask_nodes[w];
  float4 xa = *(const float4*)(x + node * DIN + lane * 4);
  float4 aa = *(const float4*)(agg2 + node * DIN + lane * 4);
  float rx = leaky(aa.x, ACT_SLOPE), ry = leaky(aa.y, ACT_SLOPE);
  float rz = leaky(aa.z, ACT_SLOPE), rw = leaky(aa.w, ACT_SLOPE);
  float dot = rx*xa.x + ry*xa.y + rz*xa.z + rw*xa.w;
  float rr  = rx*rx + ry*ry + rz*rz + rw*rw;
  float xx  = xa.x*xa.x + xa.y*xa.y + xa.z*xa.z + xa.w*xa.w;
#pragma unroll
  for (int off = 16; off > 0; off >>= 1) {
    dot += __shfl_xor(dot, off, 32);
    rr  += __shfl_xor(rr, off, 32);
    xx  += __shfl_xor(xx, off, 32);
  }
  if (lane == 0) {
    float na = fmaxf(sqrtf(rr), 1e-8f);
    float nb = fmaxf(sqrtf(xx), 1e-8f);
    atomicAdd(acc, dot / (na * nb));
  }
}

__global__ void k_final(const float* __restrict__ acc, float* __restrict__ out, int NM) {
  if (threadIdx.x == 0 && blockIdx.x == 0) out[0] = 1.f - acc[0] / (float)NM;
}

// ---------------- launcher ----------------
extern "C" void kernel_launch(void* const* d_in, const int* in_sizes, int n_in,
                              void* d_out, int out_size, void* d_ws, size_t ws_size,
                              hipStream_t stream) {
  const float* x         = (const float*)d_in[0];
  const int*   ei        = (const int*)d_in[1];
  const int*   mask      = (const int*)d_in[2];
  const float* token     = (const float*)d_in[3];
  const float* encW      = (const float*)d_in[4];
  const float* enc_as    = (const float*)d_in[5];
  const float* enc_ad    = (const float*)d_in[6];
  const float* enc_b     = (const float*)d_in[7];
  const float* e2dW      = (const float*)d_in[8];
  const float* decW      = (const float*)d_in[9];
  const float* dec_as    = (const float*)d_in[10];
  const float* dec_ad    = (const float*)d_in[11];
  const float* dec_b     = (const float*)d_in[12];
  float* out = (float*)d_out;

  const int N  = in_sizes[0] / DIN;
  const int E  = in_sizes[1] / 2;
  const int NM = in_sizes[2];
  const int ET = E + N;
  const int NT = (N + 15) / 16;

  auto rnd = [](size_t v) { return (v + 255) & ~(size_t)255; };
  float* ws = (float*)d_ws;
  size_t o = 0;
  float* henc  = ws + o; o += rnd((size_t)N * DH);   // h_enc, later reused as rep
  float* agg1  = ws + o; o += rnd((size_t)N * DH);   // encoder aggregate -> enc_rep
  float* asrc  = ws + o; o += rnd((size_t)N);
  float* adst  = ws + o; o += rnd((size_t)N);
  float* mbuf  = ws + o; o += rnd((size_t)N);
  float* den   = ws + o; o += rnd((size_t)N);
  int*   mflag = (int*)(ws + o); o += rnd((size_t)N);
  float* ebuf  = ws + o; o += rnd((size_t)ET);
  float* agg2  = ws + o; o += rnd((size_t)N * DIN);
  float* wsrc  = ws + o;            // 8 floats
  float* wdst  = ws + o + 8;        // 8 floats
  float* acc   = ws + o + 16;       // 1 float

  const int B = 256;
  k_init<<<(N * DIN + B - 1) / B, B, 0, stream>>>(agg2, dec_b, agg1, enc_b, mbuf, den, mflag, acc, N);
  k_scatter_mask<<<(NM + B - 1) / B, B, 0, stream>>>(mask, mflag, NM);
  k_enc_gemm<<<(NT * 32 + B - 1) / B, B, 0, stream>>>(x, token, mflag, encW, henc, N, NT);
  k_enc_alpha<<<(N + B - 1) / B, B, 0, stream>>>(henc, enc_as, enc_ad, asrc, adst, N);
  k_edge_logits<<<(ET + B - 1) / B, B, 0, stream>>>(ei, asrc, adst, mbuf, ebuf, E, ET);
  k_edge_exp<<<(ET + B - 1) / B, B, 0, stream>>>(ei, mbuf, den, ebuf, E, ET);
  k_edge_agg_enc<<<(ET + B - 1) / B, B, 0, stream>>>(ei, henc, ebuf, den, agg1, E, ET);
  k_enc_act<<<(N * DH + B - 1) / B, B, 0, stream>>>(agg1, out + 1, N * DH);
  k_fold<<<1, 16, 0, stream>>>(decW, dec_as, dec_ad, wsrc, wdst);
  k_e2d<<<(N + B - 1) / B, B, 0, stream>>>(agg1, e2dW, mflag, wsrc, wdst, henc, asrc, adst, N);
  k_reinit<<<(N + B - 1) / B, B, 0, stream>>>(mbuf, den, N);
  k_edge_logits<<<(ET + B - 1) / B, B, 0, stream>>>(ei, asrc, adst, mbuf, ebuf, E, ET);
  k_edge_exp<<<(ET + B - 1) / B, B, 0, stream>>>(ei, mbuf, den, ebuf, E, ET);
  k_edge_agg_dec<<<((size_t)ET * 32 + B - 1) / B, B, 0, stream>>>(ei, henc, decW, ebuf, den, agg2, E, ET);
  k_loss<<<(NM * 32 + B - 1) / B, B, 0, stream>>>(mask, x, agg2, acc, NM);
  k_final<<<1, 1, 0, stream>>>(acc, out, NM);
  (void)n_in; (void)out_size; (void)ws_size;
}
